// SelfAttentionModule_65481071410829
// MI455X (gfx1250) — matrix operand order repaired and checked by reading, not compiled, not run on hardware
//
#include <hip/hip_runtime.h>

// ---------------------------------------------------------------------------
// Self-attention for MI455X (gfx1250, wave32, WMMA).
//   Q = X @ Wq ; K = X @ Wk ; O = softmax(Q K^T / 32) @ X
// N = 8192 tokens, d = 1024.
//
// Strategy (see analysis): bf16 operands + f32 accumulation via
// v_wmma_f32_16x16x32_bf16. All operand matrices (68 MB bf16) are L2-resident
// (192 MB L2), so fragments are loaded directly from global memory as two
// contiguous 16B per-lane chunks; B-side matrices are pre-transposed once so
// the WMMA B-fragment layout is also contiguous. Flash-attention (online
// softmax) keeps O in registers and avoids the 256 MB score matrix.
//
// Workspace layout (bf16 = u16 elements), total 68 MB:
//   Xb  [8192][1024]  @ 0
//   XbT [1024][8192]  @ 8M      (V^T for the PV GEMM)
//   Qb  [8192][1024]  @ 16M
//   Kb  [8192][1024]  @ 24M
//   WqT [1024][1024]  @ 32M
//   WkT [1024][1024]  @ 33M
// ---------------------------------------------------------------------------

#define N_TOK 8192
#define EDIM  1024
#define SCALE 0.03125f   // 1/sqrt(1024)

typedef __attribute__((ext_vector_type(4)))  unsigned int v4u;
typedef __attribute__((ext_vector_type(8)))  unsigned int v8u;
typedef __attribute__((ext_vector_type(8)))  float        v8f;
typedef __attribute__((ext_vector_type(16))) __bf16       v16bf;

// f32 -> bf16 with round-to-nearest-even
static __device__ __forceinline__ unsigned short f32_bf16(float f) {
  unsigned u = __float_as_uint(f);
  return (unsigned short)((u + 0x7fffu + ((u >> 16) & 1u)) >> 16);
}
static __device__ __forceinline__ unsigned int pack2_bf16(float x, float y) {
  return (unsigned)f32_bf16(x) | ((unsigned)f32_bf16(y) << 16);
}

// Build a 16xbf16 WMMA fragment from two contiguous 16B chunks.
// (Matches the CDNA5 16-bit A/B VGPR layouts when p0/p1 are chosen per-lane.)
static __device__ __forceinline__ v16bf frag2(const unsigned short* p0,
                                              const unsigned short* p1) {
  v4u lo = *reinterpret_cast<const v4u*>(p0);
  v4u hi = *reinterpret_cast<const v4u*>(p1);
  v8u u;
  u[0] = lo[0]; u[1] = lo[1]; u[2] = lo[2]; u[3] = lo[3];
  u[4] = hi[0]; u[5] = hi[1]; u[6] = hi[2]; u[7] = hi[3];
  return __builtin_bit_cast(v16bf, u);
}

static __device__ __forceinline__ v8f wmma_bf16(v16bf a, v16bf b, v8f c) {
  // (neg_a, A, neg_b, B, c_mod, C, reuse_a, reuse_b)
  return __builtin_amdgcn_wmma_f32_16x16x32_bf16(false, a, false, b,
                                                 (short)0, c, false, false);
}

// --------------------------- conversion kernels ----------------------------

// X (f32 row-major) -> Xb (bf16 row-major) and XbT (bf16 transposed)
__global__ __launch_bounds__(256) void cvt_x(const float* __restrict__ x,
                                             unsigned short* __restrict__ xb,
                                             unsigned short* __restrict__ xbt) {
  int i = blockIdx.x * 256 + threadIdx.x;            // one element, grid = N*D/256
  int r = i >> 10, c = i & 1023;
  unsigned short v = f32_bf16(x[i]);
  xb[i] = v;
  xbt[(size_t)c * N_TOK + r] = v;
}

// W (f32 [k][n] row-major) -> WT (bf16 [n][k])
__global__ __launch_bounds__(256) void cvt_wt(const float* __restrict__ w,
                                              unsigned short* __restrict__ wt) {
  int i = blockIdx.x * 256 + threadIdx.x;            // grid = D*D/256
  int r = i >> 10, c = i & 1023;
  wt[(size_t)c * EDIM + r] = f32_bf16(w[i]);
}

// ------------------------- projection GEMM (Q/K) ---------------------------
// C[m][n] = sum_k A[m][k] * WT[n][k], bf16 in, bf16 out, f32 accumulate.
// Block tile 64(M) x 256(N), 8 waves in a 2x4 grid, 32x64 per wave.
__global__ __launch_bounds__(256) void proj_gemm(
    const unsigned short* __restrict__ A,   // [N_TOK][EDIM]
    const unsigned short* __restrict__ WT,  // [EDIM(n)][EDIM(k)]
    unsigned short* __restrict__ C) {       // [N_TOK][EDIM]
  const int m0 = blockIdx.x * 64, n0 = blockIdx.y * 256;
  const int w = threadIdx.x >> 5, l = threadIdx.x & 31;
  const int ll = l & 15, lh = l >> 4;
  const int wm = w & 1, wn = w >> 1;

  v8f acc[2][4];
#pragma unroll
  for (int r = 0; r < 2; ++r)
#pragma unroll
    for (int c = 0; c < 4; ++c)
#pragma unroll
      for (int i = 0; i < 8; ++i) acc[r][c][i] = 0.0f;

#pragma unroll 2
  for (int kk = 0; kk < EDIM; kk += 32) {
    v16bf a[2], b[4];
#pragma unroll
    for (int r = 0; r < 2; ++r) {
      const unsigned short* p =
          A + (size_t)(m0 + 32 * wm + 16 * r + ll) * EDIM + kk + 8 * lh;
      a[r] = frag2(p, p + 16);                 // A: K 0..7 / 16..23 per half
    }
#pragma unroll
    for (int c = 0; c < 4; ++c) {
      const unsigned short* p =
          WT + (size_t)(n0 + 64 * wn + 16 * c + ll) * EDIM + kk + 8 * lh;
      b[c] = frag2(p, p + 16);                 // B: K 0..7 / 16..23 per half
    }
#pragma unroll
    for (int r = 0; r < 2; ++r)
#pragma unroll
      for (int c = 0; c < 4; ++c) acc[r][c] = wmma_bf16(a[r], b[c], acc[r][c]);
  }

#pragma unroll
  for (int r = 0; r < 2; ++r)
#pragma unroll
    for (int c = 0; c < 4; ++c)
#pragma unroll
      for (int i = 0; i < 8; ++i) {
        int row = m0 + 32 * wm + 16 * r + i + 8 * lh;
        int col = n0 + 64 * wn + 16 * c + ll;
        C[(size_t)row * EDIM + col] = f32_bf16(acc[r][c][i]);
      }
}

// ------------------------- flash attention kernel --------------------------
// Block: 32 query rows, 256 threads = 8 waves (om = w&1 row group, on = w>>1
// column quarter). Iterate over 128 KV blocks of 64 keys. O stays in VGPRs.
__global__ __launch_bounds__(256) void attn(
    const unsigned short* __restrict__ Qb,   // [N_TOK][EDIM]
    const unsigned short* __restrict__ Kb,   // [N_TOK][EDIM]
    const unsigned short* __restrict__ XbT,  // [EDIM][N_TOK]  (V transposed)
    float* __restrict__ out) {               // [N_TOK][EDIM] f32
  __shared__ float ldsS[32][68];             // 32x64 score tile (+pad)
  __shared__ float mLds[32], lLds[32], aLds[32];

  const int q0 = blockIdx.x * 32;
  const int w = threadIdx.x >> 5, l = threadIdx.x & 31;
  const int ll = l & 15, lh = l >> 4;
  const int om = w & 1, on = w >> 1;         // O rows 16*om.., cols 256*on..

  if (threadIdx.x < 32) { mLds[threadIdx.x] = -3.0e38f; lLds[threadIdx.x] = 0.0f; }

  v8f o[16];
#pragma unroll
  for (int nt = 0; nt < 16; ++nt)
#pragma unroll
    for (int i = 0; i < 8; ++i) o[nt][i] = 0.0f;

  __syncthreads();

  const unsigned short* qp = Qb + (size_t)(q0 + 16 * om + ll) * EDIM + 8 * lh;

  for (int it = 0; it < N_TOK / 64; ++it) {
    const int kv0 = it * 64;
    if (it + 1 < N_TOK / 64)                       // global_prefetch_b8 of next K tile
      __builtin_prefetch(Kb + (size_t)(kv0 + 64) * EDIM, 0, 1);

    // ---- S tile: rows 16*om.., cols kv0 + 16*on..  (full K=1024 reduction)
    v8f s;
#pragma unroll
    for (int i = 0; i < 8; ++i) s[i] = 0.0f;
    const unsigned short* kp =
        Kb + (size_t)(kv0 + 16 * on + ll) * EDIM + 8 * lh;
#pragma unroll 4
    for (int kk = 0; kk < EDIM; kk += 32) {
      v16bf a = frag2(qp + kk, qp + kk + 16);
      v16bf b = frag2(kp + kk, kp + kk + 16);      // B = K^T row n contiguous
      s = wmma_bf16(a, b, s);
    }
#pragma unroll
    for (int i = 0; i < 8; ++i)
      ldsS[16 * om + i + 8 * lh][16 * on + ll] = s[i] * SCALE;
    __syncthreads();

    // ---- online softmax stats, one thread per query row
    if (threadIdx.x < 32) {
      const int r = threadIdx.x;
      float mOld = mLds[r], rm = mOld;
      for (int j = 0; j < 64; ++j) rm = fmaxf(rm, ldsS[r][j]);
      float alpha = __expf(mOld - rm);
      float sum = 0.0f;
      for (int j = 0; j < 64; ++j) {
        float p = __expf(ldsS[r][j] - rm);
        ldsS[r][j] = p;                            // P tile in place
        sum += p;
      }
      mLds[r] = rm;
      aLds[r] = alpha;
      lLds[r] = lLds[r] * alpha + sum;
    }
    __syncthreads();

    // ---- rescale running O by alpha
    float al[8];
#pragma unroll
    for (int i = 0; i < 8; ++i) al[i] = aLds[16 * om + i + 8 * lh];
#pragma unroll
    for (int nt = 0; nt < 16; ++nt)
#pragma unroll
      for (int i = 0; i < 8; ++i) o[nt][i] *= al[i];

    // ---- O += P @ V  (V = X, fragments from XbT so K-pairs are contiguous)
#pragma unroll
    for (int ks = 0; ks < 2; ++ks) {
      const int kk = 32 * ks;
      v8u pu;
#pragma unroll
      for (int i = 0; i < 8; ++i) {
        int k = kk + 2 * (i & 3) + ((i >> 2) << 4) + 8 * lh;  // A-frag layout
        pu[i] = pack2_bf16(ldsS[16 * om + ll][k], ldsS[16 * om + ll][k + 1]);
      }
      v16bf pa = __builtin_bit_cast(v16bf, pu);
#pragma unroll
      for (int nt = 0; nt < 16; ++nt) {
        const unsigned short* vp =
            XbT + (size_t)(256 * on + 16 * nt + ll) * N_TOK + kv0 + kk + 8 * lh;
        v16bf vb = frag2(vp, vp + 16);
        o[nt] = wmma_bf16(pa, vb, o[nt]);
      }
    }
    __syncthreads();                               // ldsS reused next iteration
  }

  // ---- epilogue: divide by row sums, store f32
  float li[8];
#pragma unroll
  for (int i = 0; i < 8; ++i) li[i] = 1.0f / lLds[16 * om + i + 8 * lh];
#pragma unroll
  for (int nt = 0; nt < 16; ++nt)
#pragma unroll
    for (int i = 0; i < 8; ++i) {
      int row = q0 + 16 * om + i + 8 * lh;
      int col = 256 * on + 16 * nt + ll;
      out[(size_t)row * EDIM + col] = o[nt][i] * li[i];
    }
}

// ------------------------------- launcher ----------------------------------
extern "C" void kernel_launch(void* const* d_in, const int* in_sizes, int n_in,
                              void* d_out, int out_size, void* d_ws, size_t ws_size,
                              hipStream_t stream) {
  (void)in_sizes; (void)n_in; (void)out_size; (void)ws_size;  // needs >= 68 MB ws
  const float* Wq = (const float*)d_in[0];   // rotation_params
  const float* Wk = (const float*)d_in[1];   // entangle_params
  const float* X  = (const float*)d_in[2];   // inputs

  unsigned short* ws  = (unsigned short*)d_ws;
  const size_t M = (size_t)1024 * 1024;
  unsigned short* Xb  = ws;
  unsigned short* XbT = ws + 8 * M;
  unsigned short* Qb  = ws + 16 * M;
  unsigned short* Kb  = ws + 24 * M;
  unsigned short* WqT = ws + 32 * M;
  unsigned short* WkT = ws + 33 * M;

  cvt_x <<<(N_TOK * EDIM) / 256, 256, 0, stream>>>(X, Xb, XbT);
  cvt_wt<<<(EDIM * EDIM) / 256, 256, 0, stream>>>(Wq, WqT);
  cvt_wt<<<(EDIM * EDIM) / 256, 256, 0, stream>>>(Wk, WkT);

  dim3 pg(N_TOK / 64, EDIM / 256);
  proj_gemm<<<pg, 256, 0, stream>>>(Xb, WqT, Qb);
  proj_gemm<<<pg, 256, 0, stream>>>(Xb, WkT, Kb);

  attn<<<N_TOK / 32, 256, 0, stream>>>(Qb, Kb, XbT, (float*)d_out);
}
